// EGNNLayer_1168231105096
// MI455X (gfx1250) — matrix operand order repaired and checked by reading, not compiled, not run on hardware
//
#include <hip/hip_runtime.h>
#include <hip/hip_bf16.h>
#include <math.h>

typedef __attribute__((ext_vector_type(16))) __bf16 v16bf;
typedef __attribute__((ext_vector_type(8)))  float  v8f;

union FragU { unsigned u[8]; v16bf v; };

// ---------- small helpers ----------

__device__ __forceinline__ unsigned pack2_bf16(float a, float b) {
#if __has_builtin(__builtin_amdgcn_cvt_pk_bf16_f32)
  auto t = __builtin_amdgcn_cvt_pk_bf16_f32(a, b);
  unsigned r; __builtin_memcpy(&r, &t, sizeof(r));
  return r;
#else
  unsigned ua = __float_as_uint(a) + 0x8000u;
  unsigned ub = __float_as_uint(b) + 0x8000u;
  return __builtin_amdgcn_perm(ub, ua, 0x07060302u);
#endif
}

__device__ __forceinline__ unsigned short bf16_1(float a) {
  return (unsigned short)((__float_as_uint(a) + 0x8000u) >> 16);
}

__device__ __forceinline__ v8f wmma_bf16(v16bf a, v16bf b, v8f c) {
  return __builtin_amdgcn_wmma_f32_16x16x32_bf16(false, a, false, b,
                                                 (short)0, c, false, false);
}

template<int ACT>  // 0 = identity, 1 = silu, 2 = tanh
__device__ __forceinline__ float activate(float v) {
  if (ACT == 1) return v * __builtin_amdgcn_rcpf(1.0f + __expf(-v));
  if (ACT == 2) return __builtin_fmaf(-2.0f,
                  __builtin_amdgcn_rcpf(__expf(2.0f * v) + 1.0f), 1.0f);
  return v;
}

// A fragment (16x32 bf16) from a per-lane row of 128 bf16 (64 dwords).
__device__ __forceinline__ v16bf afrag_b16(const unsigned* row, int ktsub, int lane) {
  const int kb4 = (lane >> 4) << 2;
  const uint4* p0 = reinterpret_cast<const uint4*>(row + ktsub * 16 + kb4);
  const uint4* p1 = reinterpret_cast<const uint4*>(row + ktsub * 16 + 8 + kb4);
  uint4 a = p0[0], b = p1[0];
  FragU f;
  f.u[0] = a.x; f.u[1] = a.y; f.u[2] = a.z; f.u[3] = a.w;
  f.u[4] = b.x; f.u[5] = b.y; f.u[6] = b.z; f.u[7] = b.w;
  return f.v;
}

// A fragment from a per-lane fp32 row (aggr rows in the node kernel).
__device__ __forceinline__ v16bf afrag_f32(const float* __restrict__ row,
                                           int ktsub, int lane) {
  const int kb = (lane >> 4) << 3;
  const float4* p0 = reinterpret_cast<const float4*>(row + ktsub * 32 + kb);
  const float4* p1 = reinterpret_cast<const float4*>(row + ktsub * 32 + 16 + kb);
  float4 a = p0[0], b = p0[1], c = p1[0], d = p1[1];
  FragU f;
  f.u[0] = pack2_bf16(a.x, a.y); f.u[1] = pack2_bf16(a.z, a.w);
  f.u[2] = pack2_bf16(b.x, b.y); f.u[3] = pack2_bf16(b.z, b.w);
  f.u[4] = pack2_bf16(c.x, c.y); f.u[5] = pack2_bf16(c.z, c.w);
  f.u[6] = pack2_bf16(d.x, d.y); f.u[7] = pack2_bf16(d.z, d.w);
  return f.v;
}

// B fragment (32x16 bf16) from pre-swizzled weights: 8 contiguous dwords/lane.
__device__ __forceinline__ v16bf bfrag(const unsigned* __restrict__ w,
                                       int kt, int nt, int lane) {
  const uint4* p = reinterpret_cast<const uint4*>(w + ((kt * 8 + nt) << 8) + (lane << 3));
  uint4 a = p[0], b = p[1];
  FragU f;
  f.u[0] = a.x; f.u[1] = a.y; f.u[2] = a.z; f.u[3] = a.w;
  f.u[4] = b.x; f.u[5] = b.y; f.u[6] = b.z; f.u[7] = b.w;
  return f.v;
}

__device__ __forceinline__ void gatomic_fadd(float* p, float v) {
  asm volatile("global_atomic_add_f32 %0, %1, off" :: "v"(p), "v"(v) : "memory");
}

// ---------- setup kernels ----------

__global__ void zero_kernel(float* __restrict__ p, long long n) {
  long long i = (long long)blockIdx.x * 256 + threadIdx.x;
  if (i < n) p[i] = 0.0f;
}

__global__ void pack_rows_kernel(const float* __restrict__ src,
                                 unsigned* __restrict__ dst, long long ndw) {
  long long i = (long long)blockIdx.x * 256 + threadIdx.x;
  if (i >= ndw) return;
  float2 v = reinterpret_cast<const float2*>(src)[i];
  dst[i] = pack2_bf16(v.x, v.y);
}

// fp32 row-major [K x 128] -> bf16 WMMA-B fragment order, ktiles K/32 tiles.
__global__ void swizzle_w(const float* __restrict__ W, unsigned* __restrict__ dst,
                          int ktiles) {
  int d = blockIdx.x * 256 + threadIdx.x;
  int total = ktiles * 8 * 256;
  if (d >= total) return;
  int tile = d >> 8, within = d & 255;
  int lane = within >> 3, v = within & 7;
  int kt = tile >> 3, nt = tile & 7;
  int n = lane & 15, khalf = lane >> 4;
  int k = kt * 32 + khalf * 8 + 2 * (v & 3) + 16 * (v >> 2);
  float a = W[(long long)k * 128 + nt * 16 + n];
  float b = W[(long long)(k + 1) * 128 + nt * 16 + n];
  dst[d] = pack2_bf16(a, b);
}

// ---------- edge-side: 2-layer MLP on a 32-edge (2 x 16) tile ----------
// Two M-tiles share every B fragment -> half the weight traffic per edge.

template<int ACT>
__device__ __forceinline__ void edge_mlp2(
    const unsigned* r0A0, const unsigned* r0A1,   // sub-tile 0 rows (K lo / hi)
    const unsigned* r1A0, const unsigned* r1A1,   // sub-tile 1 rows
    const unsigned* __restrict__ W1s, const unsigned* __restrict__ W2s,
    const float* __restrict__ w1last,             // fp32 row 256 (dist weights)
    const float* __restrict__ b1, const float* __restrict__ b2,
    const float* dm0, const float* dm1, const int* rc0, const int* rc1,
    unsigned short (*hb0)[128], unsigned short (*hb1)[128],
    const unsigned* hrd0, const unsigned* hrd1,
    int lane, float* __restrict__ aggrOut) {
  const int m  = lane & 15;
  const int hi = (lane >> 4) << 3;
  v8f acc0[8] = {}, acc1[8] = {};
#pragma unroll
  for (int kt = 0; kt < 8; kt++) {
    v16bf a0 = afrag_b16(kt < 4 ? r0A0 : r0A1, kt & 3, lane);
    v16bf a1 = afrag_b16(kt < 4 ? r1A0 : r1A1, kt & 3, lane);
#pragma unroll
    for (int nt = 0; nt < 8; nt++) {
      v16bf b = bfrag(W1s, kt, nt, lane);
      acc0[nt] = wmma_bf16(a0, b, acc0[nt]);
      acc1[nt] = wmma_bf16(a1, b, acc1[nt]);
    }
  }
#pragma unroll
  for (int nt = 0; nt < 8; nt++) {
    float wl = w1last[nt * 16 + m];
    float bb = b1[nt * 16 + m];
#pragma unroll
    for (int r = 0; r < 8; r++) {
      hb0[hi + r][nt * 16 + m] =
          bf16_1(activate<ACT>(acc0[nt][r] + dm0[r] * wl + bb));
      hb1[hi + r][nt * 16 + m] =
          bf16_1(activate<ACT>(acc1[nt][r] + dm1[r] * wl + bb));
    }
  }
  asm volatile("s_wait_dscnt 0" ::: "memory");
  v16bf a20[4], a21[4];
#pragma unroll
  for (int kt = 0; kt < 4; kt++) {
    a20[kt] = afrag_b16(hrd0, kt, lane);
    a21[kt] = afrag_b16(hrd1, kt, lane);
  }
#pragma unroll
  for (int nt = 0; nt < 8; nt++) {
    v8f c0 = {}, c1 = {};
#pragma unroll
    for (int kt = 0; kt < 4; kt++) {
      v16bf b = bfrag(W2s, kt, nt, lane);
      c0 = wmma_bf16(a20[kt], b, c0);
      c1 = wmma_bf16(a21[kt], b, c1);
    }
    float bb = b2[nt * 16 + m];
#pragma unroll
    for (int r = 0; r < 8; r++) {
      gatomic_fadd(aggrOut + 128ll * rc0[r] + nt * 16 + m,
                   activate<ACT>(c0[r] + bb));
      gatomic_fadd(aggrOut + 128ll * rc1[r] + nt * 16 + m,
                   activate<ACT>(c1[r] + bb));
    }
  }
}

__global__ __launch_bounds__(256) void egnn_edge_kernel(
    const unsigned* __restrict__ xb, const unsigned* __restrict__ peb,
    const float* __restrict__ pos, const long long* __restrict__ eidx,
    const float* __restrict__ mm_w1, const float* __restrict__ mm_b1,
    const float* __restrict__ mm_b2,
    const float* __restrict__ mp_w1, const float* __restrict__ mp_b1,
    const float* __restrict__ mp_b2,
    const unsigned* __restrict__ mmW1s, const unsigned* __restrict__ mmW2s,
    const unsigned* __restrict__ mpW1s, const unsigned* __restrict__ mpW2s,
    float* __restrict__ aggr, float* __restrict__ aggrPos, int E) {
  __shared__ unsigned short hb[8][2][16][128];  // 64 KB: per-wave double hidden tile
  const int lane = threadIdx.x & 31;
  const int wave = threadIdx.x >> 5;
  const long long tile = (long long)blockIdx.x * 8 + wave;
  if (tile * 32 >= E) return;  // wave-uniform: EXEC stays all-ones

  const int m  = lane & 15;
  const int hi = (lane >> 4) << 3;
  const int e0 = (int)(tile * 32) + m;       // sub-tile 0 edge for this lane
  const int e1 = e0 + 16;                    // sub-tile 1 edge
  const long long s0 = eidx[e0], r0 = eidx[(long long)E + e0];
  const long long s1 = eidx[e1], r1 = eidx[(long long)E + e1];
  float dx0 = pos[3*s0]-pos[3*r0], dy0 = pos[3*s0+1]-pos[3*r0+1], dz0 = pos[3*s0+2]-pos[3*r0+2];
  float dx1 = pos[3*s1]-pos[3*r1], dy1 = pos[3*s1+1]-pos[3*r1+1], dz1 = pos[3*s1+2]-pos[3*r1+2];
  const float dist0 = sqrtf(dx0*dx0 + dy0*dy0 + dz0*dz0);
  const float dist1 = sqrtf(dx1*dx1 + dy1*dy1 + dz1*dz1);

  float dm0[8], dm1[8]; int rc0[8], rc1[8];
#pragma unroll
  for (int r = 0; r < 8; r++) {
    dm0[r] = __shfl(dist0, hi + r);  rc0[r] = __shfl((int)r0, hi + r);
    dm1[r] = __shfl(dist1, hi + r);  rc1[r] = __shfl((int)r1, hi + r);
  }

  const unsigned* row0x  = xb  + 64ll * r0;
  const unsigned* row0pr = peb + 64ll * r0;
  const unsigned* row0ps = peb + 64ll * s0;
  const unsigned* row1x  = xb  + 64ll * r1;
  const unsigned* row1pr = peb + 64ll * r1;
  const unsigned* row1ps = peb + 64ll * s1;
  const unsigned* hrd0 = (const unsigned*)&hb[wave][0][m][0];
  const unsigned* hrd1 = (const unsigned*)&hb[wave][1][m][0];

  // message = silu/silu MLP on [x[rec], pe[rec], dist] -> atomic into aggr
  edge_mlp2<1>(row0x, row0pr, row1x, row1pr, mmW1s, mmW2s,
               mm_w1 + 256 * 128, mm_b1, mm_b2, dm0, dm1, rc0, rc1,
               hb[wave][0], hb[wave][1], hrd0, hrd1, lane, aggr);
  // message_pos = tanh/tanh MLP on [pe[send], pe[rec], dist] -> aggrPos
  edge_mlp2<2>(row0ps, row0pr, row1ps, row1pr, mpW1s, mpW2s,
               mp_w1 + 256 * 128, mp_b1, mp_b2, dm0, dm1, rc0, rc1,
               hb[wave][0], hb[wave][1], hrd0, hrd1, lane, aggrPos);
}

// ---------- node-side: 2-layer MLP on a 16-node tile, dense store ----------

template<int KT1, int KTB, int ACTH, int ACTO>
__device__ __forceinline__ void node_mlp(
    const unsigned* rb0, const unsigned* rb1, const float* rf,
    const unsigned* __restrict__ W1s, const unsigned* __restrict__ W2s,
    const float* __restrict__ b1, const float* __restrict__ b2,
    unsigned short (*hbw)[128], const unsigned* hrd,
    int lane, float* __restrict__ outBase, long long nodeBase) {
  const int m  = lane & 15;
  const int hi = (lane >> 4) << 3;
  v8f acc[8] = {};
#pragma unroll
  for (int kt = 0; kt < KT1; kt++) {
    v16bf a = (kt < KTB) ? afrag_b16(kt < 4 ? rb0 : rb1, kt & 3, lane)
                         : afrag_f32(rf, kt & 3, lane);
#pragma unroll
    for (int nt = 0; nt < 8; nt++)
      acc[nt] = wmma_bf16(a, bfrag(W1s, kt, nt, lane), acc[nt]);
  }
#pragma unroll
  for (int nt = 0; nt < 8; nt++) {
    float bb = b1[nt * 16 + m];
#pragma unroll
    for (int r = 0; r < 8; r++)
      hbw[hi + r][nt * 16 + m] = bf16_1(activate<ACTH>(acc[nt][r] + bb));
  }
  asm volatile("s_wait_dscnt 0" ::: "memory");
  v16bf a2[4];
#pragma unroll
  for (int kt = 0; kt < 4; kt++) a2[kt] = afrag_b16(hrd, kt, lane);
#pragma unroll
  for (int nt = 0; nt < 8; nt++) {
    v8f c = {};
#pragma unroll
    for (int kt = 0; kt < 4; kt++)
      c = wmma_bf16(a2[kt], bfrag(W2s, kt, nt, lane), c);
    float bb = b2[nt * 16 + m];
#pragma unroll
    for (int r = 0; r < 8; r++)
      outBase[128ll * (nodeBase + hi + r) + nt * 16 + m] = activate<ACTO>(c[r] + bb);
  }
}

__global__ __launch_bounds__(256) void egnn_node_kernel(
    const unsigned* __restrict__ xb, const unsigned* __restrict__ peb,
    const float* __restrict__ aggr, const float* __restrict__ aggrPos,
    const unsigned* __restrict__ upW1s, const unsigned* __restrict__ upW2s,
    const unsigned* __restrict__ ueW1s, const unsigned* __restrict__ ueW2s,
    const float* __restrict__ up_b1, const float* __restrict__ up_b2,
    const float* __restrict__ ue_b1, const float* __restrict__ ue_b2,
    float* __restrict__ outUpd, float* __restrict__ outPe, int Nn) {
  __shared__ unsigned short hb[8][16][128];
  const int lane = threadIdx.x & 31;
  const int wave = threadIdx.x >> 5;
  const long long tile = (long long)blockIdx.x * 8 + wave;
  if (tile * 16 >= Nn) return;  // wave-uniform guard

  const int m = lane & 15;
  const long long node = tile * 16 + m;
  const unsigned* rowxb  = xb  + 64ll * node;
  const unsigned* rowpeb = peb + 64ll * node;
  const float* rowag  = aggr    + 128ll * node;
  const float* rowagp = aggrPos + 128ll * node;
  const unsigned* hrd = (const unsigned*)&hb[wave][m][0];

  node_mlp<12, 8, 1, 0>(rowxb, rowpeb, rowag, upW1s, upW2s, up_b1, up_b2,
                        hb[wave], hrd, lane, outUpd, tile * 16);
  node_mlp<8, 4, 2, 2>(rowpeb, rowpeb, rowagp, ueW1s, ueW2s, ue_b1, ue_b2,
                       hb[wave], hrd, lane, outPe, tile * 16);
}

// ---------- host launcher ----------

extern "C" void kernel_launch(void* const* d_in, const int* in_sizes, int n_in,
                              void* d_out, int out_size, void* d_ws, size_t ws_size,
                              hipStream_t stream) {
  const float* x      = (const float*)d_in[0];
  const float* pos    = (const float*)d_in[1];
  const long long* ei = (const long long*)d_in[2];
  const float* pe     = (const float*)d_in[3];
  const float* mm_w1 = (const float*)d_in[4];  const float* mm_b1 = (const float*)d_in[5];
  const float* mm_w2 = (const float*)d_in[6];  const float* mm_b2 = (const float*)d_in[7];
  const float* mp_w1 = (const float*)d_in[8];  const float* mp_b1 = (const float*)d_in[9];
  const float* mp_w2 = (const float*)d_in[10]; const float* mp_b2 = (const float*)d_in[11];
  const float* up_w1 = (const float*)d_in[12]; const float* up_b1 = (const float*)d_in[13];
  const float* up_w2 = (const float*)d_in[14]; const float* up_b2 = (const float*)d_in[15];
  const float* ue_w1 = (const float*)d_in[16]; const float* ue_b1 = (const float*)d_in[17];
  const float* ue_w2 = (const float*)d_in[18]; const float* ue_b2 = (const float*)d_in[19];

  const int Nn = in_sizes[0] / 128;  // 50000
  const int E  = in_sizes[2] / 2;    // 320000

  // workspace layout
  char* ws = (char*)d_ws;
  float* aggr    = (float*)ws; ws += (size_t)Nn * 128 * sizeof(float);
  float* aggrPos = (float*)ws; ws += (size_t)Nn * 128 * sizeof(float);
  unsigned* xbv  = (unsigned*)ws; ws += (size_t)Nn * 64 * 4;
  unsigned* pebv = (unsigned*)ws; ws += (size_t)Nn * 64 * 4;
  unsigned* mmW1s = (unsigned*)ws; ws += 16384 * 4;
  unsigned* mpW1s = (unsigned*)ws; ws += 16384 * 4;
  unsigned* mmW2s = (unsigned*)ws; ws += 8192  * 4;
  unsigned* mpW2s = (unsigned*)ws; ws += 8192  * 4;
  unsigned* upW1s = (unsigned*)ws; ws += 24576 * 4;
  unsigned* upW2s = (unsigned*)ws; ws += 8192  * 4;
  unsigned* ueW1s = (unsigned*)ws; ws += 16384 * 4;
  unsigned* ueW2s = (unsigned*)ws; ws += 8192  * 4;

  long long zn = 2ll * Nn * 128;
  zero_kernel<<<(unsigned)((zn + 255) / 256), 256, 0, stream>>>(aggr, zn);

  long long ndw = (long long)Nn * 64;
  pack_rows_kernel<<<(unsigned)((ndw + 255) / 256), 256, 0, stream>>>(x,  xbv,  ndw);
  pack_rows_kernel<<<(unsigned)((ndw + 255) / 256), 256, 0, stream>>>(pe, pebv, ndw);

  swizzle_w<<<(8  * 2048 + 255) / 256, 256, 0, stream>>>(mm_w1, mmW1s, 8);
  swizzle_w<<<(8  * 2048 + 255) / 256, 256, 0, stream>>>(mp_w1, mpW1s, 8);
  swizzle_w<<<(4  * 2048 + 255) / 256, 256, 0, stream>>>(mm_w2, mmW2s, 4);
  swizzle_w<<<(4  * 2048 + 255) / 256, 256, 0, stream>>>(mp_w2, mpW2s, 4);
  swizzle_w<<<(12 * 2048 + 255) / 256, 256, 0, stream>>>(up_w1, upW1s, 12);
  swizzle_w<<<(4  * 2048 + 255) / 256, 256, 0, stream>>>(up_w2, upW2s, 4);
  swizzle_w<<<(8  * 2048 + 255) / 256, 256, 0, stream>>>(ue_w1, ueW1s, 8);
  swizzle_w<<<(4  * 2048 + 255) / 256, 256, 0, stream>>>(ue_w2, ueW2s, 4);

  int etiles = (E + 31) / 32;   // 10000 32-edge tiles
  egnn_edge_kernel<<<(etiles + 7) / 8, 256, 0, stream>>>(
      xbv, pebv, pos, ei, mm_w1, mm_b1, mm_b2, mp_w1, mp_b1, mp_b2,
      mmW1s, mmW2s, mpW1s, mpW2s, aggr, aggrPos, E);

  int ntiles = (Nn + 15) / 16;  // 3125
  float* outUpd = (float*)d_out;
  float* outPe  = outUpd + (size_t)Nn * 128;
  egnn_node_kernel<<<(ntiles + 7) / 8, 256, 0, stream>>>(
      xbv, pebv, aggr, aggrPos, upW1s, upW2s, ueW1s, ueW2s,
      up_b1, up_b2, ue_b1, ue_b2, outUpd, outPe, Nn);
}
